// LocalAttention_84069689852514
// MI455X (gfx1250) — matrix operand order repaired and checked by reading, not compiled, not run on hardware
//
#include <hip/hip_runtime.h>
#include <hip/hip_bf16.h>

#define BATCH 32
#define UAVS  512
#define TGTS  2048
#define EMB   256
#define HDIM  32
#define KNN   16

typedef __attribute__((ext_vector_type(16))) __bf16 v16bf;
typedef __attribute__((ext_vector_type(8)))  __bf16 v8bf;
typedef __attribute__((ext_vector_type(4)))  __bf16 v4bf;
typedef __attribute__((ext_vector_type(8)))  float  v8f;

#define CAT16(lo, hi) __builtin_shufflevector((lo), (hi), 0, 1, 2, 3, 4, 5, 6, 7, \
                                              8, 9, 10, 11, 12, 13, 14, 15)

// ---------------------------------------------------------------------------
// Kernel 0: streaming f32 -> bf16 conversion (done ONCE per operand, so the
// GEMM hot loops are pure bf16 b128 loads + wmma, no per-use v_cvt).
// ---------------------------------------------------------------------------
__global__ __launch_bounds__(256) void cvt_f32_to_bf16(const float* __restrict__ in,
                                                       __bf16* __restrict__ out,
                                                       int n4) {
  const int i = blockIdx.x * 256 + threadIdx.x;
  if (i < n4) {
    const float4 f = ((const float4*)in)[i];
    v4bf o;
    o[0] = (__bf16)f.x; o[1] = (__bf16)f.y; o[2] = (__bf16)f.z; o[3] = (__bf16)f.w;
    ((v4bf*)out)[i] = o;
  }
}

// ---------------------------------------------------------------------------
// Kernel 1: per-query 16-smallest-distance selection (one wave per row).
// ---------------------------------------------------------------------------
__global__ __launch_bounds__(32) void knn_topk16(const float* __restrict__ dist,
                                                 int* __restrict__ sel) {
  const int row  = blockIdx.x;               // b*UAVS + u
  const int lane = threadIdx.x;
  const float* d = dist + (size_t)row * TGTS;

  float v[TGTS / 32];
#pragma unroll
  for (int i = 0; i < TGTS / 32; ++i) v[i] = d[i * 32 + lane];

#pragma unroll 1
  for (int s = 0; s < KNN; ++s) {
    float best = 3.402823e38f;
    int   bi   = 0x7fffffff;
#pragma unroll
    for (int i = 0; i < TGTS / 32; ++i) {
      const int idx = i * 32 + lane;
      const bool better = (v[i] < best) || (v[i] == best && idx < bi);
      best = better ? v[i] : best;
      bi   = better ? idx  : bi;
    }
#pragma unroll
    for (int m = 16; m > 0; m >>= 1) {
      const float ob = __shfl_xor(best, m, 32);
      const int   oi = __shfl_xor(bi,   m, 32);
      const bool better = (ob < best) || (ob == best && oi < bi);
      best = better ? ob : best;
      bi   = better ? oi : bi;
    }
    if (lane == 0) sel[row * KNN + s] = bi;
#pragma unroll
    for (int i = 0; i < TGTS / 32; ++i)
      if (bi == i * 32 + lane) v[i] = 3.402823e38f;   // winner is wave-uniform
  }
}

// ---------------------------------------------------------------------------
// GEMM  out[m,n] = sum_e A[m,e] * W[n,e]  (+ optional bias), E = 256,
// all-bf16 operands, f32 accumulate. One wave per 16(M) x 64(N) tile:
// one A fragment feeds 4 v_wmma_f32_16x16x32_bf16 per 32-wide K slab.
// Fragment layouts (A/B: contiguous k order used here; g = lane/16, l16 = lane%16):
//   A: lane = 16g + m ; VGPRs hold K = [16g..16g+15]                  (2x16B)
//   B: lane = 16g + n ; VGPRs hold K = [16g..16g+15]                  (2x16B)
//   D: col = l16 ; VGPR r: row = r + 8g
// ---------------------------------------------------------------------------
template <bool OUT_BF16, bool HAS_BIAS>
__global__ __launch_bounds__(256) void gemm_bf16_e256(const __bf16* __restrict__ A,
                                                      const __bf16* __restrict__ W,
                                                      const float* __restrict__ bias,
                                                      void* __restrict__ Outp,
                                                      int Mtiles) {
  const int wave = threadIdx.x >> 5;
  const int lane = threadIdx.x & 31;
  const int tile = blockIdx.x * 8 + wave;   // tile = 16 rows x 64 cols
  const int mt   = tile >> 2;
  const int ng   = tile & 3;                // N-group: cols [ng*64, ng*64+64)
  if (mt >= Mtiles) return;                 // wave-uniform branch

  const int g   = lane >> 4;
  const int l16 = lane & 15;

  const __bf16* Arow = A + (size_t)(mt * 16 + l16) * EMB;
  const __bf16* Wr0  = W + (size_t)(ng * 64 +  0 + l16) * EMB;
  const __bf16* Wr1  = W + (size_t)(ng * 64 + 16 + l16) * EMB;
  const __bf16* Wr2  = W + (size_t)(ng * 64 + 32 + l16) * EMB;
  const __bf16* Wr3  = W + (size_t)(ng * 64 + 48 + l16) * EMB;

  v8f c0 = {}, c1 = {}, c2 = {}, c3 = {};
#pragma unroll
  for (int kt = 0; kt < EMB / 32; ++kt) {
    const int k0 = kt * 32;
    if (kt + 2 < EMB / 32)
      __builtin_prefetch(Arow + k0 + 64, 0, 3);   // global_prefetch next A slab

    const v8bf alo = *(const v8bf*)(Arow + k0 + 16 * g);
    const v8bf ahi = *(const v8bf*)(Arow + k0 + 16 * g + 8);
    const v16bf a  = CAT16(alo, ahi);

    const v16bf b0 = CAT16(*(const v8bf*)(Wr0 + k0 + 16 * g),
                           *(const v8bf*)(Wr0 + k0 + 16 * g + 8));
    const v16bf b1 = CAT16(*(const v8bf*)(Wr1 + k0 + 16 * g),
                           *(const v8bf*)(Wr1 + k0 + 16 * g + 8));
    const v16bf b2 = CAT16(*(const v8bf*)(Wr2 + k0 + 16 * g),
                           *(const v8bf*)(Wr2 + k0 + 16 * g + 8));
    const v16bf b3 = CAT16(*(const v8bf*)(Wr3 + k0 + 16 * g),
                           *(const v8bf*)(Wr3 + k0 + 16 * g + 8));

    c0 = __builtin_amdgcn_wmma_f32_16x16x32_bf16(false, a, false, b0, (short)0, c0, false, false);
    c1 = __builtin_amdgcn_wmma_f32_16x16x32_bf16(false, a, false, b1, (short)0, c1, false, false);
    c2 = __builtin_amdgcn_wmma_f32_16x16x32_bf16(false, a, false, b2, (short)0, c2, false, false);
    c3 = __builtin_amdgcn_wmma_f32_16x16x32_bf16(false, a, false, b3, (short)0, c3, false, false);
  }

#pragma unroll
  for (int sub = 0; sub < 4; ++sub) {
    const v8f& c = (sub == 0) ? c0 : (sub == 1) ? c1 : (sub == 2) ? c2 : c3;
    const int col = ng * 64 + sub * 16 + l16;
    const float bv = HAS_BIAS ? bias[col] : 0.0f;
#pragma unroll
    for (int r = 0; r < 8; ++r) {
      const int rowi = mt * 16 + r + 8 * g;
      const float o  = c[r] + bv;
      if (OUT_BF16)
        ((__bf16*)Outp)[(size_t)rowi * EMB + col] = (__bf16)o;
      else
        ((float*)Outp)[(size_t)rowi * EMB + col] = o;
    }
  }
}

// ---------------------------------------------------------------------------
// Kernel 5: per-query 16-neighbor attention. One block per (b,u), one wave
// per head; K/V gathers are L2-resident (2 MB per batch).
// ---------------------------------------------------------------------------
__global__ __launch_bounds__(256) void local_attn_softmax(
    const float* __restrict__ Q, const __bf16* __restrict__ Kall,
    const __bf16* __restrict__ Vall, const int* __restrict__ sel,
    __bf16* __restrict__ attn_out) {
  const int bu   = blockIdx.x;        // b*UAVS + u
  const int b    = bu >> 9;           // UAVS == 512
  const int head = threadIdx.x >> 5;  // wave id == head
  const int lane = threadIdx.x & 31;

  const float qd = Q[(size_t)bu * EMB + head * HDIM + lane];  // q[dim=lane]

  const __bf16* Kb = Kall + (size_t)b * TGTS * EMB;
  const __bf16* Vb = Vall + (size_t)b * TGTS * EMB;
  const int* idxp  = sel + bu * KNN;

  const int j16   = lane & 15;        // my neighbor (mirrored in upper half)
  const int myidx = idxp[j16];

  // score_j = (1/sqrt(32)) * sum_d q[d] * K[idx_j][head*32 + d]
  const __bf16* krow = Kb + (size_t)myidx * EMB + head * HDIM;
  float s = 0.0f;
#pragma unroll
  for (int d = 0; d < HDIM; ++d) {
    const float qv = __shfl(qd, d, 32);
    s += qv * (float)krow[d];
  }
  s *= 0.17677669529663687f;          // 1/sqrt(32), TEMPERATURE = 1

  float mx = s;
#pragma unroll
  for (int off = 8; off > 0; off >>= 1) mx = fmaxf(mx, __shfl_xor(mx, off, 32));
  const float e = __expf(s - mx);
  float sum = e;
#pragma unroll
  for (int off = 8; off > 0; off >>= 1) sum += __shfl_xor(sum, off, 32);
  const float a = e / sum;

  float acc = 0.0f;
#pragma unroll
  for (int j = 0; j < KNN; ++j) {
    const float aj = __shfl(a, j, 32);
    const int   ij = __shfl(myidx, j, 32);
    acc += aj * (float)Vb[(size_t)ij * EMB + head * HDIM + lane];
  }
  attn_out[(size_t)bu * EMB + head * HDIM + lane] = (__bf16)acc;
}

// ---------------------------------------------------------------------------
// Launcher. Inputs: uav[0], tgt[1], dist[2], Wq[3], Wk[4], Wv[5], Wo[6], bo[7]
// ---------------------------------------------------------------------------
extern "C" void kernel_launch(void* const* d_in, const int* in_sizes, int n_in,
                              void* d_out, int out_size, void* d_ws, size_t ws_size,
                              hipStream_t stream) {
  (void)in_sizes; (void)n_in; (void)out_size; (void)ws_size;
  const float* uav  = (const float*)d_in[0];
  const float* tgt  = (const float*)d_in[1];
  const float* dist = (const float*)d_in[2];
  const float* Wq   = (const float*)d_in[3];
  const float* Wk   = (const float*)d_in[4];
  const float* Wv   = (const float*)d_in[5];
  const float* Wo   = (const float*)d_in[6];
  const float* bo   = (const float*)d_in[7];
  float* out = (float*)d_out;

  char* ws = (char*)d_ws;
  size_t off = 0;
  auto alloc = [&](size_t bytes) -> void* {
    void* p = ws + off;
    off = (off + bytes + 255) & ~(size_t)255;
    return p;
  };
  const size_t nUav = (size_t)BATCH * UAVS * EMB;   // 4.19M
  const size_t nTgt = (size_t)BATCH * TGTS * EMB;   // 16.8M
  const size_t nW   = (size_t)EMB * EMB;            // 65536

  int*    sel   = (int*)   alloc((size_t)BATCH * UAVS * KNN * sizeof(int));
  float*  Q     = (float*) alloc(nUav * sizeof(float));
  __bf16* Kall  = (__bf16*)alloc(nTgt * sizeof(__bf16));
  __bf16* Vall  = (__bf16*)alloc(nTgt * sizeof(__bf16));
  __bf16* attnO = (__bf16*)alloc(nUav * sizeof(__bf16));
  __bf16* uavB  = (__bf16*)alloc(nUav * sizeof(__bf16));
  __bf16* tgtB  = (__bf16*)alloc(nTgt * sizeof(__bf16));
  __bf16* WqB   = (__bf16*)alloc(nW * sizeof(__bf16));
  __bf16* WkB   = (__bf16*)alloc(nW * sizeof(__bf16));
  __bf16* WvB   = (__bf16*)alloc(nW * sizeof(__bf16));
  __bf16* WoB   = (__bf16*)alloc(nW * sizeof(__bf16));

  // one-time f32 -> bf16 staging
  cvt_f32_to_bf16<<<(int)(nUav / 4 + 255) / 256, 256, 0, stream>>>(uav, uavB, (int)(nUav / 4));
  cvt_f32_to_bf16<<<(int)(nTgt / 4 + 255) / 256, 256, 0, stream>>>(tgt, tgtB, (int)(nTgt / 4));
  cvt_f32_to_bf16<<<(int)(nW / 4 + 255) / 256, 256, 0, stream>>>(Wq, WqB, (int)(nW / 4));
  cvt_f32_to_bf16<<<(int)(nW / 4 + 255) / 256, 256, 0, stream>>>(Wk, WkB, (int)(nW / 4));
  cvt_f32_to_bf16<<<(int)(nW / 4 + 255) / 256, 256, 0, stream>>>(Wv, WvB, (int)(nW / 4));
  cvt_f32_to_bf16<<<(int)(nW / 4 + 255) / 256, 256, 0, stream>>>(Wo, WoB, (int)(nW / 4));

  knn_topk16<<<BATCH * UAVS, 32, 0, stream>>>(dist, sel);

  const int MtQ = BATCH * UAVS / 16;   // 1024 M-tiles (16384 rows)
  const int MtT = BATCH * TGTS / 16;   // 4096 M-tiles (65536 rows)

  // Q = uav @ Wq^T (f32 out); project ALL targets once: K/V = tgt @ W^T (bf16)
  gemm_bf16_e256<false, false><<<MtQ * 4 / 8, 256, 0, stream>>>(uavB, WqB, nullptr, Q,    MtQ);
  gemm_bf16_e256<true,  false><<<MtT * 4 / 8, 256, 0, stream>>>(tgtB, WkB, nullptr, Kall, MtT);
  gemm_bf16_e256<true,  false><<<MtT * 4 / 8, 256, 0, stream>>>(tgtB, WvB, nullptr, Vall, MtT);

  local_attn_softmax<<<BATCH * UAVS, 256, 0, stream>>>(Q, Kall, Vall, sel, attnO);

  // out = attnO @ Wo^T + bo (f32 out)
  gemm_bf16_e256<false, true><<<MtQ * 4 / 8, 256, 0, stream>>>(attnO, WoB, bo, out, MtQ);
}
